// JTNNDecoder_35158602285573
// MI455X (gfx1250) — matrix operand order repaired and checked by reading, not compiled, not run on hardware
//
#include <hip/hip_runtime.h>
#include <cstdint>
#include <cstddef>

// Problem constants (match reference)
#define N_    16384
#define MAXNB 15
#define H_    512
#define L_    128
#define V_    1024

typedef __attribute__((ext_vector_type(16))) __bf16 v16bf;
typedef __attribute__((ext_vector_type(8)))  float  v8f;
typedef unsigned short bf16_t;   // raw bf16 storage

union BF16Frag { v16bf v; uint4 q[2]; };

__device__ __forceinline__ bf16_t f2bf(float f) {
  unsigned u = __builtin_bit_cast(unsigned, f);
  u += 0x7fffu + ((u >> 16) & 1u);   // round-to-nearest-even
  return (bf16_t)(u >> 16);
}
__device__ __forceinline__ float bf2f(bf16_t h) {
  unsigned u = ((unsigned)h) << 16;
  return __builtin_bit_cast(float, u);
}
__device__ __forceinline__ void store_bf4(bf16_t* dst, float a, float b, float c, float d) {
  uint2 u;
  u.x = (unsigned)f2bf(a) | ((unsigned)f2bf(b) << 16);
  u.y = (unsigned)f2bf(c) | ((unsigned)f2bf(d) << 16);
  *reinterpret_cast<uint2*>(dst) = u;
}

__device__ __forceinline__ v8f wmma_bf16(v16bf a, v16bf b, v8f c) {
  return __builtin_amdgcn_wmma_f32_16x16x32_bf16(false, a, false, b, (short)0, c,
                                                 false, false);
}

// ---- CDNA5 async copy to LDS (ASYNCcnt path, ISA 15.18.3 op 98) ------------
__device__ __forceinline__ void async_load_b128(unsigned lds_byte_off,
                                                const bf16_t* gptr) {
  asm volatile("global_load_async_to_lds_b128 %0, %1, off"
               :
               : "v"(lds_byte_off), "v"(gptr)
               : "memory");
}
__device__ __forceinline__ void wait_asynccnt0() {
#if __has_builtin(__builtin_amdgcn_s_wait_asynccnt)
  __builtin_amdgcn_s_wait_asynccnt(0);
#else
  asm volatile("s_wait_asynccnt 0" ::: "memory");
#endif
}

// ---------------------------------------------------------------------------
// fp32 -> bf16 weight conversion
// ---------------------------------------------------------------------------
__global__ __launch_bounds__(256) void convert_f32_to_bf16(
    const float* __restrict__ src, bf16_t* __restrict__ dst, int n) {
  int i = blockIdx.x * blockDim.x + threadIdx.x;
  int stride = gridDim.x * blockDim.x;
  for (; i < n; i += stride) dst[i] = f2bf(src[i]);
}

// ---------------------------------------------------------------------------
// Prep: gathers, masked neighbor sums, bf16 packing of all GEMM A-matrices.
// ---------------------------------------------------------------------------
__global__ __launch_bounds__(128) void prep_kernel(
    const int* __restrict__ x_ids, const int* __restrict__ contexts,
    const int* __restrict__ nei_idx, const int* __restrict__ nei_mask,
    const int* __restrict__ o_idx, const int* __restrict__ o_mask,
    const float* __restrict__ H_msg, const float* __restrict__ x_tree,
    const float* __restrict__ embedding,
    bf16_t* __restrict__ Az,    // [N,1024] = x || sum_h
    bf16_t* __restrict__ Aio,   // [N,1024] = x || cur_o
    bf16_t* __restrict__ Ah,    // [N,1024] = x || sum_gated(later)
    bf16_t* __restrict__ Aw,    // [N,640]  = new_h(later) || ctx
    bf16_t* __restrict__ As,    // [N,640]  = stop_h(later) || ctx
    bf16_t* __restrict__ hnei,  // [N*15,512]
    float*  __restrict__ sumh)  // [N,512]
{
  const int n = blockIdx.x;
  const int t = threadIdx.x;
  const int d = t * 4;

  const float4 x4 = *reinterpret_cast<const float4*>(
      embedding + (size_t)x_ids[n] * H_ + d);

  float sh0 = 0.f, sh1 = 0.f, sh2 = 0.f, sh3 = 0.f;
  float co0 = 0.f, co1 = 0.f, co2 = 0.f, co3 = 0.f;

#pragma unroll 1
  for (int k = 0; k < MAXNB; ++k) {
    const int   mi = nei_idx[n * MAXNB + k];
    const float mm = (float)nei_mask[n * MAXNB + k];
    const float4 h4 = *reinterpret_cast<const float4*>(H_msg + (size_t)mi * H_ + d);
    const float a = h4.x * mm, b = h4.y * mm, c = h4.z * mm, e = h4.w * mm;
    sh0 += a; sh1 += b; sh2 += c; sh3 += e;
    store_bf4(hnei + ((size_t)n * MAXNB + k) * H_ + d, a, b, c, e);

    const int   oi = o_idx[n * MAXNB + k];
    const float om = (float)o_mask[n * MAXNB + k];
    const float4 o4 = *reinterpret_cast<const float4*>(H_msg + (size_t)oi * H_ + d);
    co0 += o4.x * om; co1 += o4.y * om; co2 += o4.z * om; co3 += o4.w * om;
  }

  *reinterpret_cast<float4*>(sumh + (size_t)n * H_ + d) =
      make_float4(sh0, sh1, sh2, sh3);

  store_bf4(Az  + (size_t)n * 1024 + d,       x4.x, x4.y, x4.z, x4.w);
  store_bf4(Az  + (size_t)n * 1024 + 512 + d, sh0, sh1, sh2, sh3);
  store_bf4(Aio + (size_t)n * 1024 + d,       x4.x, x4.y, x4.z, x4.w);
  store_bf4(Aio + (size_t)n * 1024 + 512 + d, co0, co1, co2, co3);
  store_bf4(Ah  + (size_t)n * 1024 + d,       x4.x, x4.y, x4.z, x4.w);

  if (t < L_ / 4) {
    const float4 c4 = *reinterpret_cast<const float4*>(
        x_tree + (size_t)contexts[n] * L_ + d);
    store_bf4(Aw + (size_t)n * 640 + 512 + d, c4.x, c4.y, c4.z, c4.w);
    store_bf4(As + (size_t)n * 640 + 512 + d, c4.x, c4.y, c4.z, c4.w);
  }
}

// ---------------------------------------------------------------------------
// Generic bf16 WMMA GEMM: C[M,Nc] = act(A[M,K] * Bw[Nc,K]^T + bias)
// 256 threads = 8 waves; wave = 16 rows x 64 cols. Block = 32 rows x 256 cols.
// B is staged per 32-K chunk into LDS via GLOBAL_LOAD_ASYNC_TO_LDS_B128
// (double-buffered, padded stride 40 elems to avoid bank conflicts).
// Requires M%32==0, Nc%256==0, K%32==0.
// ---------------------------------------------------------------------------
enum { ACT_NONE = 0, ACT_SIGMOID = 1, ACT_RELU = 2 };
#define BPAD 40   // padded per-col stride (elements) in LDS chunk

template <int ACT, bool STORE_BF, bool GRU>
__global__ __launch_bounds__(256) void gemm_bf16_kernel(
    const bf16_t* __restrict__ A, int lda,
    const bf16_t* __restrict__ Bw, int K,
    const float* __restrict__ bias,
    void* __restrict__ Cout, int ldc,
    const float* __restrict__ zf,     // [N,512] (GRU only)
    const float* __restrict__ sumhf)  // [N,512] (GRU only)
{
  __shared__ bf16_t Bsh[2][256 * BPAD];   // 2 x 20KB

  const int t    = threadIdx.x;
  const int l    = t & 31;
  const int w    = t >> 5;
  const int lrow = l & 15;
  const int lhi  = l >> 4;
  const int r0   = blockIdx.y * 32 + (w >> 2) * 16;
  const int cblk = blockIdx.x * 256;
  const int c0l  = (w & 3) * 64;

  auto stage = [&](int buf, int kk) {
#pragma unroll
    for (int i = 0; i < 4; ++i) {
      const int idx = t + 256 * i;       // 1024 x 16B transfers = 32K chunk
      const int col = idx >> 2;
      const int kq  = idx & 3;
      const unsigned ldsOff =
          (unsigned)(size_t)&Bsh[buf][col * BPAD + kq * 8];
      async_load_b128(ldsOff, Bw + (size_t)(cblk + col) * K + kk + kq * 8);
    }
  };

  v8f acc[4];
#pragma unroll
  for (int j = 0; j < 4; ++j)
#pragma unroll
    for (int i = 0; i < 8; ++i) acc[j][i] = 0.0f;

  const bf16_t* aRow = A + (size_t)(r0 + lrow) * lda;
  const int nsteps = K >> 5;

  stage(0, 0);
#pragma unroll 1
  for (int s = 0; s < nsteps; ++s) {
    const int kk = s << 5;
    wait_asynccnt0();
    __syncthreads();
    if (s + 1 < nsteps) stage((s + 1) & 1, kk + 32);
    __builtin_prefetch(aRow + kk + 128, 0, 0);  // global_prefetch_b8 ahead of A stream

    BF16Frag af;   // A frag: lane=row; K split {0..7,16..23}/{8..15,24..31}
    af.q[0] = *reinterpret_cast<const uint4*>(aRow + kk + lhi * 8);
    af.q[1] = *reinterpret_cast<const uint4*>(aRow + kk + 16 + lhi * 8);

    const bf16_t* bbase = Bsh[s & 1];
#pragma unroll
    for (int j = 0; j < 4; ++j) {        // B frag from LDS: lane=col
      const bf16_t* bRow = bbase + (c0l + j * 16 + lrow) * BPAD + lhi * 8;
      BF16Frag bfr;
      bfr.q[0] = *reinterpret_cast<const uint4*>(bRow);
      bfr.q[1] = *reinterpret_cast<const uint4*>(bRow + 16);
      acc[j] = wmma_bf16(af.v, bfr.v, acc[j]);
    }
    __syncthreads();
  }

#pragma unroll
  for (int j = 0; j < 4; ++j) {
    const int col = cblk + c0l + j * 16 + lrow;
    const float bv = bias[col];
#pragma unroll
    for (int i = 0; i < 8; ++i) {
      const int row = r0 + i + lhi * 8;   // C layout: VGPR i -> M=i / i+8
      float v = acc[j][i] + bv;
      if (GRU) {
        const float zz = zf[(size_t)row * H_ + col];
        const float sv = sumhf[(size_t)row * H_ + col];
        v = (1.0f - zz) * sv + zz * tanhf(v);
      } else if (ACT == ACT_SIGMOID) {
        v = 1.0f / (1.0f + __expf(-v));
      } else if (ACT == ACT_RELU) {
        v = fmaxf(v, 0.0f);
      }
      if (STORE_BF)
        reinterpret_cast<bf16_t*>(Cout)[(size_t)row * ldc + col] = f2bf(v);
      else
        reinterpret_cast<float*>(Cout)[(size_t)row * ldc + col] = v;
    }
  }
}

// ---------------------------------------------------------------------------
// Fused r2 einsum + gate + reduce:
//   for k in 0..14: r2 = h_nei[:,k,:] @ Ur^T ; sum_gated += sigmoid(r1+r2)*h_nei
// Block = 128 n-rows x 64 cols (8 waves, one 16-row tile each, shared cols).
// The 64-col x 512-K Ur strip (64KB) is async-staged into LDS ONCE, with an
// XOR group swizzle, then consumed 15x from LDS (no register hoist/spill).
// ---------------------------------------------------------------------------
__global__ __launch_bounds__(256) void r2_gated_kernel(
    const bf16_t* __restrict__ hnei,  // [N*15,512]
    const bf16_t* __restrict__ Urb,   // [512,512]
    const float*  __restrict__ r1,    // [N,512]
    bf16_t* __restrict__ Ah)          // [N,1024]
{
  __shared__ bf16_t Ush[64 * 512];    // 64KB, group-swizzled

  const int t    = threadIdx.x;
  const int l    = t & 31;
  const int w    = t >> 5;
  const int lrow = l & 15;
  const int lhi  = l >> 4;
  const int n0   = blockIdx.x * 128;
  const int c0   = blockIdx.y * 64;
  const int r0   = n0 + w * 16;

  // Stage Ur[c0:c0+64, 0:512] -> LDS. 4096 x 16B async transfers.
#pragma unroll
  for (int i = 0; i < 16; ++i) {
    const int idx = t + 256 * i;        // 0..4095
    const int col = idx >> 6;           // 0..63
    const int g   = idx & 63;           // 8-element K group
    const int gs  = g ^ (col & 15);     // bank swizzle
    const unsigned ldsOff = (unsigned)(size_t)&Ush[col * 512 + gs * 8];
    async_load_b128(ldsOff, Urb + (size_t)(c0 + col) * 512 + g * 8);
  }
  wait_asynccnt0();
  __syncthreads();

  float r1v[4][8];
  v8f gated[4];
#pragma unroll
  for (int j = 0; j < 4; ++j)
#pragma unroll
    for (int i = 0; i < 8; ++i) {
      const int row = r0 + i + lhi * 8;
      const int col = c0 + j * 16 + lrow;
      r1v[j][i] = r1[(size_t)row * H_ + col];
      gated[j][i] = 0.0f;
    }

#pragma unroll 1
  for (int knb = 0; knb < MAXNB; ++knb) {
    v8f acc[4];
#pragma unroll
    for (int j = 0; j < 4; ++j)
#pragma unroll
      for (int i = 0; i < 8; ++i) acc[j][i] = 0.0f;

    const bf16_t* aRow = hnei + ((size_t)(r0 + lrow) * MAXNB + knb) * H_;
#pragma unroll 2
    for (int kk = 0; kk < H_; kk += 32) {
      __builtin_prefetch(aRow + kk + 128, 0, 0);
      BF16Frag af;
      af.q[0] = *reinterpret_cast<const uint4*>(aRow + kk + lhi * 8);
      af.q[1] = *reinterpret_cast<const uint4*>(aRow + kk + 16 + lhi * 8);
      const int gbase = (kk >> 3) + lhi;       // this lane's first K group
#pragma unroll
      for (int j = 0; j < 4; ++j) {
        const int col = j * 16 + lrow;
        const int sw  = col & 15;
        BF16Frag bfr;
        bfr.q[0] = *reinterpret_cast<const uint4*>(
            &Ush[col * 512 + ((gbase)     ^ sw) * 8]);
        bfr.q[1] = *reinterpret_cast<const uint4*>(
            &Ush[col * 512 + ((gbase + 2) ^ sw) * 8]);
        acc[j] = wmma_bf16(af.v, bfr.v, acc[j]);
      }
    }

#pragma unroll
    for (int j = 0; j < 4; ++j)
#pragma unroll
      for (int i = 0; i < 8; ++i) {
        const int row = r0 + i + lhi * 8;
        const int col = c0 + j * 16 + lrow;
        const float s = 1.0f / (1.0f + __expf(-(r1v[j][i] + acc[j][i])));
        const float hv = bf2f(hnei[((size_t)row * MAXNB + knb) * H_ + col]);
        gated[j][i] += s * hv;
      }
  }

#pragma unroll
  for (int j = 0; j < 4; ++j)
#pragma unroll
    for (int i = 0; i < 8; ++i) {
      const int row = r0 + i + lhi * 8;
      const int col = c0 + j * 16 + lrow;
      Ah[(size_t)row * 1024 + 512 + col] = f2bf(gated[j][i]);
    }
}

// ---------------------------------------------------------------------------
// Stop score: out[n,1024] = s_hid[n,:] . Uo_w + Uo_b   (Nc = 1 GEMV)
// ---------------------------------------------------------------------------
__global__ __launch_bounds__(256) void stop_score_kernel(
    const bf16_t* __restrict__ shid, const float* __restrict__ Uo_w,
    const float* __restrict__ Uo_b, float* __restrict__ out) {
  const int n = blockIdx.x * blockDim.x + threadIdx.x;
  if (n >= N_) return;
  float acc = Uo_b[0];
  const bf16_t* row = shid + (size_t)n * H_;
#pragma unroll 4
  for (int k = 0; k < H_; ++k) acc += bf2f(row[k]) * Uo_w[k];
  out[(size_t)n * (V_ + 1) + V_] = acc;
}

// ---------------------------------------------------------------------------
extern "C" void kernel_launch(void* const* d_in, const int* in_sizes, int n_in,
                              void* d_out, int out_size, void* d_ws,
                              size_t ws_size, hipStream_t stream) {
  const int*   x_ids     = (const int*)d_in[0];
  const int*   contexts  = (const int*)d_in[1];
  const int*   nei_idx   = (const int*)d_in[2];
  const int*   nei_mask  = (const int*)d_in[3];
  const int*   o_idx     = (const int*)d_in[4];
  const int*   o_mask    = (const int*)d_in[5];
  const float* H_msg     = (const float*)d_in[6];
  const float* x_tree    = (const float*)d_in[7];
  const float* embedding = (const float*)d_in[8];
  const float* Wz_w = (const float*)d_in[9];
  const float* Wz_b = (const float*)d_in[10];
  const float* Wr_w = (const float*)d_in[11];
  const float* Wr_b = (const float*)d_in[12];
  const float* Ur_w = (const float*)d_in[13];
  const float* Wh_w = (const float*)d_in[14];
  const float* Wh_b = (const float*)d_in[15];
  const float* W_w  = (const float*)d_in[16];
  const float* W_b  = (const float*)d_in[17];
  const float* Wo_w = (const float*)d_in[18];
  const float* Wo_b = (const float*)d_in[19];
  const float* Ui_w = (const float*)d_in[20];
  const float* Ui_b = (const float*)d_in[21];
  const float* U_w  = (const float*)d_in[22];
  const float* U_b  = (const float*)d_in[23];
  const float* Uo_w = (const float*)d_in[24];
  const float* Uo_b = (const float*)d_in[25];
  float* out = (float*)d_out;

  // ---- workspace carve-out ----
  char* ws = (char*)d_ws;
  size_t off = 0;
  auto alloc = [&](size_t bytes) -> void* {
    void* p = ws + off;
    off += (bytes + 255) & ~(size_t)255;
    return p;
  };
  const size_t Nz = (size_t)N_;
  bf16_t* Az   = (bf16_t*)alloc(Nz * 1024 * 2);
  bf16_t* Aio  = (bf16_t*)alloc(Nz * 1024 * 2);
  bf16_t* Ah   = (bf16_t*)alloc(Nz * 1024 * 2);
  bf16_t* Aw   = (bf16_t*)alloc(Nz * 640 * 2);
  bf16_t* As   = (bf16_t*)alloc(Nz * 640 * 2);
  bf16_t* hnei = (bf16_t*)alloc(Nz * MAXNB * H_ * 2);
  float*  sumh = (float*)alloc(Nz * H_ * 4);
  float*  z_f  = (float*)alloc(Nz * H_ * 4);
  float*  r1_f = (float*)alloc(Nz * H_ * 4);
  bf16_t* whid = (bf16_t*)alloc(Nz * H_ * 2);
  bf16_t* shid = (bf16_t*)alloc(Nz * H_ * 2);
  bf16_t* Wzb = (bf16_t*)alloc((size_t)512 * 1024 * 2);
  bf16_t* Wrb = (bf16_t*)alloc((size_t)512 * 512 * 2);
  bf16_t* Urb = (bf16_t*)alloc((size_t)512 * 512 * 2);
  bf16_t* Whb = (bf16_t*)alloc((size_t)512 * 1024 * 2);
  bf16_t* Wb  = (bf16_t*)alloc((size_t)512 * 640 * 2);
  bf16_t* Wob = (bf16_t*)alloc((size_t)1024 * 512 * 2);
  bf16_t* Uib = (bf16_t*)alloc((size_t)512 * 1024 * 2);
  bf16_t* Ub  = (bf16_t*)alloc((size_t)512 * 640 * 2);
  if (off > ws_size) return;  // insufficient scratch: do nothing (deterministic)

  auto conv = [&](const float* s, bf16_t* d, int n) {
    convert_f32_to_bf16<<<(n + 1023) / 1024, 256, 0, stream>>>(s, d, n);
  };
  conv(Wz_w, Wzb, 512 * 1024);
  conv(Wr_w, Wrb, 512 * 512);
  conv(Ur_w, Urb, 512 * 512);
  conv(Wh_w, Whb, 512 * 1024);
  conv(W_w,  Wb,  512 * 640);
  conv(Wo_w, Wob, 1024 * 512);
  conv(Ui_w, Uib, 512 * 1024);
  conv(U_w,  Ub,  512 * 640);

  prep_kernel<<<N_, 128, 0, stream>>>(x_ids, contexts, nei_idx, nei_mask, o_idx,
                                      o_mask, H_msg, x_tree, embedding, Az, Aio,
                                      Ah, Aw, As, hnei, sumh);

  const dim3 blk(256);
  const dim3 g512(512 / 256, N_ / 32);   // Nc=512
  const dim3 g1024(1024 / 256, N_ / 32); // Nc=1024

  // z = sigmoid([x||sum_h] @ Wz^T + b)
  gemm_bf16_kernel<ACT_SIGMOID, false, false><<<g512, blk, 0, stream>>>(
      Az, 1024, Wzb, 1024, Wz_b, z_f, 512, nullptr, nullptr);
  // r1 = x @ Wr^T + b   (A = Az[:, :512] via lda=1024)
  gemm_bf16_kernel<ACT_NONE, false, false><<<g512, blk, 0, stream>>>(
      Az, 1024, Wrb, 512, Wr_b, r1_f, 512, nullptr, nullptr);
  // fused r2 einsum + gate + neighbor reduce -> Ah[:, 512:]
  r2_gated_kernel<<<dim3(N_ / 128, 512 / 64), blk, 0, stream>>>(hnei, Urb, r1_f, Ah);
  // new_h = (1-z)*sum_h + z*tanh([x||sum_gated] @ Wh^T + b) -> bf16 Aw[:, :512]
  gemm_bf16_kernel<ACT_NONE, true, true><<<g512, blk, 0, stream>>>(
      Ah, 1024, Whb, 1024, Wh_b, Aw, 640, z_f, sumh);
  // w_hid = relu([new_h||ctx] @ W^T + b)
  gemm_bf16_kernel<ACT_RELU, true, false><<<g512, blk, 0, stream>>>(
      Aw, 640, Wb, 640, W_b, whid, 512, nullptr, nullptr);
  // word_scores = w_hid @ Wo^T + b -> d_out[:, 0:1024]
  gemm_bf16_kernel<ACT_NONE, false, false><<<g1024, blk, 0, stream>>>(
      whid, 512, Wob, 512, Wo_b, out, V_ + 1, nullptr, nullptr);
  // stop_h = relu([x||cur_o] @ Ui^T + b) -> bf16 As[:, :512]
  gemm_bf16_kernel<ACT_RELU, true, false><<<g512, blk, 0, stream>>>(
      Aio, 1024, Uib, 1024, Ui_b, As, 640, nullptr, nullptr);
  // s_hid = relu([stop_h||ctx] @ U^T + b)
  gemm_bf16_kernel<ACT_RELU, true, false><<<g512, blk, 0, stream>>>(
      As, 640, Ub, 640, U_b, shid, 512, nullptr, nullptr);
  // stop_scores -> d_out[:, 1024]
  stop_score_kernel<<<(N_ + 255) / 256, blk, 0, stream>>>(shid, Uo_w, Uo_b, out);
}
